// qMoEModelBatched_67783173865797
// MI455X (gfx1250) — 8-bit matrix operand order repaired and checked by reading, compiled, not run on hardware
//
#include <hip/hip_runtime.h>
#include <stdint.h>

#define BB     8192
#define IND    1024
#define H1D    2048
#define H2D    2048
#define CD     1000
#define CPAD   1024
#define ED     8
#define RHD    512
#define QBF    127.0f
#define EPSF   1e-5f

typedef __attribute__((ext_vector_type(8))) int v8i;

__global__ void zero_kernel(float* __restrict__ p, size_t n) {
  size_t i = ((size_t)blockIdx.x * blockDim.x + threadIdx.x) * 4;
  #pragma unroll
  for (int j = 0; j < 4; ++j)
    if (i + j < n) p[i + j] = 0.0f;
}

__global__ __launch_bounds__(256) void router_h_kernel(
    const float* __restrict__ x, const float* __restrict__ Wr1,
    const float* __restrict__ br1, float* __restrict__ hf) {
  __shared__ float xs[16][17];
  __shared__ float wsm[16][17];
  int tx = threadIdx.x & 15, ty = threadIdx.x >> 4;
  int b0 = blockIdx.y * 16, j0 = blockIdx.x * 16;
  float acc = 0.0f;
  for (int k0 = 0; k0 < IND; k0 += 16) {
    xs[ty][tx]  = x  [(size_t)(b0 + ty) * IND + k0 + tx];
    wsm[ty][tx] = Wr1[(size_t)(j0 + ty) * IND + k0 + tx];
    __syncthreads();
    #pragma unroll
    for (int k = 0; k < 16; ++k) acc += xs[ty][k] * wsm[tx][k];
    __syncthreads();
  }
  hf[(size_t)(b0 + ty) * RHD + j0 + tx] = fmaxf(acc + br1[j0 + tx], 0.0f);
}

__global__ __launch_bounds__(256) void router_gate_kernel(
    const float* __restrict__ hf, const float* __restrict__ Wr2,
    const float* __restrict__ br2, float* __restrict__ probs,
    float* __restrict__ gates) {
  int lane = threadIdx.x & 31;
  int wave = threadIdx.x >> 5;
  int row  = blockIdx.x * 8 + wave;
  float acc[ED];
  #pragma unroll
  for (int e = 0; e < ED; ++e) acc[e] = 0.0f;
  const float* hp = hf + (size_t)row * RHD;
  for (int k = lane; k < RHD; k += 32) {
    float hv = hp[k];
    #pragma unroll
    for (int e = 0; e < ED; ++e) acc[e] += hv * Wr2[e * RHD + k];
  }
  #pragma unroll
  for (int e = 0; e < ED; ++e) {
    #pragma unroll
    for (int off = 16; off > 0; off >>= 1) acc[e] += __shfl_xor(acc[e], off, 32);
  }
  if (lane == 0) {
    float lg[ED], mx = -1e30f;
    #pragma unroll
    for (int e = 0; e < ED; ++e) { lg[e] = acc[e] + br2[e]; mx = fmaxf(mx, lg[e]); }
    float sum = 0.0f;
    #pragma unroll
    for (int e = 0; e < ED; ++e) { lg[e] = __expf(lg[e] - mx); sum += lg[e]; }
    float inv = 1.0f / sum;
    float p[ED];
    #pragma unroll
    for (int e = 0; e < ED; ++e) { p[e] = lg[e] * inv; probs[(size_t)row * ED + e] = p[e]; }
    int i1 = 0;
    #pragma unroll
    for (int e = 1; e < ED; ++e) if (p[e] > p[i1]) i1 = e;
    int i2 = (i1 == 0) ? 1 : 0;
    #pragma unroll
    for (int e = 0; e < ED; ++e) if (e != i1 && p[e] > p[i2]) i2 = e;
    #pragma unroll
    for (int e = 0; e < ED; ++e)
      gates[(size_t)row * ED + e] = (e == i1) ? p[i1] : ((e == i2) ? p[i2] : 0.0f);
  }
}

template <int D>
__global__ __launch_bounds__(256) void rowquant_kernel(
    const float* __restrict__ src, int8_t* __restrict__ q,
    float* __restrict__ inv_s) {
  constexpr int CH = D / 256;
  int row = blockIdx.x;
  int t   = threadIdx.x;
  const float* rp = src + (size_t)row * D + t * CH;
  float vals[CH];
  float m = 0.0f;
  #pragma unroll
  for (int i = 0; i < CH; ++i) { vals[i] = rp[i]; m = fmaxf(m, fabsf(vals[i])); }
  __shared__ float red[256];
  red[t] = m;
  __syncthreads();
  #pragma unroll
  for (int s = 128; s > 0; s >>= 1) {
    if (t < s) red[t] = fmaxf(red[t], red[t + s]);
    __syncthreads();
  }
  float mx = fmaxf(red[0], EPSF);
  float s  = QBF / mx;
  if (t == 0) inv_s[row] = mx / QBF;
  int8_t* qp = q + (size_t)row * D + t * CH;
  #pragma unroll
  for (int i = 0; i < CH; i += 4) {
    char4 c;
    c.x = (signed char)(int)fminf(fmaxf(rintf(vals[i + 0] * s), -QBF), QBF);
    c.y = (signed char)(int)fminf(fmaxf(rintf(vals[i + 1] * s), -QBF), QBF);
    c.z = (signed char)(int)fminf(fmaxf(rintf(vals[i + 2] * s), -QBF), QBF);
    c.w = (signed char)(int)fminf(fmaxf(rintf(vals[i + 3] * s), -QBF), QBF);
    *(char4*)(qp + i) = c;
  }
}

__global__ __launch_bounds__(256) void wabsmean_kernel(
    const float* __restrict__ W1, const float* __restrict__ W2,
    const float* __restrict__ W3, float* __restrict__ wscale) {
  int i = blockIdx.x;
  const float* p;
  size_t n;
  if (i < 8)       { p = W1 + (size_t)i * H1D * IND;        n = (size_t)H1D * IND; }
  else if (i < 16) { p = W2 + (size_t)(i - 8) * H2D * H1D;  n = (size_t)H2D * H1D; }
  else             { p = W3 + (size_t)(i - 16) * CD * H2D;  n = (size_t)CD * H2D; }
  float s = 0.0f;
  for (size_t j = threadIdx.x; j < n; j += 256) s += fabsf(p[j]);
  __shared__ float red[256];
  red[threadIdx.x] = s;
  __syncthreads();
  #pragma unroll
  for (int k = 128; k > 0; k >>= 1) {
    if (threadIdx.x < k) red[threadIdx.x] += red[threadIdx.x + k];
    __syncthreads();
  }
  if (threadIdx.x == 0) wscale[i] = fmaxf(red[0] / (float)n, EPSF);
}

__global__ __launch_bounds__(256) void wquant_kernel(
    const float* __restrict__ w, const float* __restrict__ wscale, int sbase,
    int8_t* __restrict__ q, size_t perExpert) {
  size_t i4 = ((size_t)blockIdx.x * 256 + threadIdx.x) * 4;
  if (i4 >= perExpert * 8) return;
  int e = (int)(i4 / perExpert);
  float inv = 1.0f / wscale[sbase + e];
  const float* wp = w + i4;
  char4 c;
  c.x = (signed char)(int)fminf(fmaxf(rintf(wp[0] * inv), -1.0f), 1.0f);
  c.y = (signed char)(int)fminf(fmaxf(rintf(wp[1] * inv), -1.0f), 1.0f);
  c.z = (signed char)(int)fminf(fmaxf(rintf(wp[2] * inv), -1.0f), 1.0f);
  c.w = (signed char)(int)fminf(fmaxf(rintf(wp[3] * inv), -1.0f), 1.0f);
  *(char4*)(q + i4) = c;
}

__global__ __launch_bounds__(256) void wquant3_kernel(
    const float* __restrict__ w, const float* __restrict__ wscale,
    int8_t* __restrict__ q) {
  size_t idx = ((size_t)blockIdx.x * 256 + threadIdx.x) * 4;
  const size_t perQ = (size_t)CPAD * H2D;
  if (idx >= perQ * 8) return;
  int e = (int)(idx / perQ);
  size_t r = idx % perQ;
  int n = (int)(r / H2D), k = (int)(r % H2D);
  char4 c; c.x = 0; c.y = 0; c.z = 0; c.w = 0;
  if (n < CD) {
    const float* wp = w + ((size_t)e * CD + n) * H2D + k;
    float inv = 1.0f / wscale[16 + e];
    c.x = (signed char)(int)fminf(fmaxf(rintf(wp[0] * inv), -1.0f), 1.0f);
    c.y = (signed char)(int)fminf(fmaxf(rintf(wp[1] * inv), -1.0f), 1.0f);
    c.z = (signed char)(int)fminf(fmaxf(rintf(wp[2] * inv), -1.0f), 1.0f);
    c.w = (signed char)(int)fminf(fmaxf(rintf(wp[3] * inv), -1.0f), 1.0f);
  }
  *(char4*)(q + idx) = c;
}

__device__ inline v8i load_fragA(const int8_t* ap) {
  int2 a01 = *(const int2*)(ap);
  int2 a23 = *(const int2*)(ap + 8);
  int2 a45 = *(const int2*)(ap + 32);
  int2 a67 = *(const int2*)(ap + 40);
  v8i A;
  A[0] = a01.x; A[1] = a01.y; A[2] = a23.x; A[3] = a23.y;
  A[4] = a45.x; A[5] = a45.y; A[6] = a67.x; A[7] = a67.y;
  return A;
}
__device__ inline v8i load_fragB(const int8_t* bp) {
  int4 blo = *(const int4*)(bp);
  int4 bhi = *(const int4*)(bp + 32);
  v8i B;
  B[0] = blo.x; B[1] = blo.y; B[2] = blo.z; B[3] = blo.w;
  B[4] = bhi.x; B[5] = bhi.y; B[6] = bhi.z; B[7] = bhi.w;
  return B;
}

__global__ __launch_bounds__(256) void gemm_iu8_relu_kernel(
    const int8_t* __restrict__ qA, const float* __restrict__ invA,
    const int8_t* __restrict__ qW, const float* __restrict__ wscale, int wsi,
    const float* __restrict__ bias, float* __restrict__ out, int N, int K) {
  int lane = threadIdx.x & 31;
  int wave = threadIdx.x >> 5;
  int h    = lane >> 4;
  int l15  = lane & 15;
  int m0   = blockIdx.y * 16;
  int n0   = blockIdx.x * (8 * 64) + wave * 64;
  if (n0 >= N) return;

  v8i acc[4];
  #pragma unroll
  for (int t = 0; t < 4; ++t)
    #pragma unroll
    for (int v = 0; v < 8; ++v) acc[t][v] = 0;

  const int8_t* abase = qA + (size_t)(m0 + l15) * K + h * 16;
  const int8_t* b0p   = qW + (size_t)(n0 +  0 + l15) * K + h * 16;
  const int8_t* b1p   = qW + (size_t)(n0 + 16 + l15) * K + h * 16;
  const int8_t* b2p   = qW + (size_t)(n0 + 32 + l15) * K + h * 16;
  const int8_t* b3p   = qW + (size_t)(n0 + 48 + l15) * K + h * 16;

  for (int kk = 0; kk < K; kk += 64) {
    v8i A = load_fragA(abase + kk);
    acc[0] = __builtin_amdgcn_wmma_i32_16x16x64_iu8(true, A, true, load_fragB(b0p + kk), acc[0], false, false);
    acc[1] = __builtin_amdgcn_wmma_i32_16x16x64_iu8(true, A, true, load_fragB(b1p + kk), acc[1], false, false);
    acc[2] = __builtin_amdgcn_wmma_i32_16x16x64_iu8(true, A, true, load_fragB(b2p + kk), acc[2], false, false);
    acc[3] = __builtin_amdgcn_wmma_i32_16x16x64_iu8(true, A, true, load_fragB(b3p + kk), acc[3], false, false);
  }

  float sw = wscale[wsi];
  float ia[8];
  #pragma unroll
  for (int v = 0; v < 8; ++v) ia[v] = invA[m0 + v + 8 * h] * sw;
  #pragma unroll
  for (int t = 0; t < 4; ++t) {
    int ncol = n0 + t * 16 + l15;
    float b = bias[ncol];
    #pragma unroll
    for (int v = 0; v < 8; ++v) {
      int row = m0 + v + 8 * h;
      float val = (float)acc[t][v] * ia[v] + b;
      out[(size_t)row * N + ncol] = fmaxf(val, 0.0f);
    }
  }
}

__global__ __launch_bounds__(256) void gemm3_iu8_gate_kernel(
    const int8_t* __restrict__ qA, const float* __restrict__ invA,
    const int8_t* __restrict__ qW, const float* __restrict__ wscale, int wsi,
    const float* __restrict__ bias, const float* __restrict__ gates, int e,
    float* __restrict__ out, int Npad, int Nreal, int K) {
  int lane = threadIdx.x & 31;
  int wave = threadIdx.x >> 5;
  int h    = lane >> 4;
  int l15  = lane & 15;
  int m0   = blockIdx.y * 16;
  int n0   = blockIdx.x * (8 * 64) + wave * 64;
  if (n0 >= Npad) return;

  v8i acc[4];
  #pragma unroll
  for (int t = 0; t < 4; ++t)
    #pragma unroll
    for (int v = 0; v < 8; ++v) acc[t][v] = 0;

  const int8_t* abase = qA + (size_t)(m0 + l15) * K + h * 16;
  const int8_t* b0p   = qW + (size_t)(n0 +  0 + l15) * K + h * 16;
  const int8_t* b1p   = qW + (size_t)(n0 + 16 + l15) * K + h * 16;
  const int8_t* b2p   = qW + (size_t)(n0 + 32 + l15) * K + h * 16;
  const int8_t* b3p   = qW + (size_t)(n0 + 48 + l15) * K + h * 16;

  for (int kk = 0; kk < K; kk += 64) {
    v8i A = load_fragA(abase + kk);
    acc[0] = __builtin_amdgcn_wmma_i32_16x16x64_iu8(true, A, true, load_fragB(b0p + kk), acc[0], false, false);
    acc[1] = __builtin_amdgcn_wmma_i32_16x16x64_iu8(true, A, true, load_fragB(b1p + kk), acc[1], false, false);
    acc[2] = __builtin_amdgcn_wmma_i32_16x16x64_iu8(true, A, true, load_fragB(b2p + kk), acc[2], false, false);
    acc[3] = __builtin_amdgcn_wmma_i32_16x16x64_iu8(true, A, true, load_fragB(b3p + kk), acc[3], false, false);
  }

  float sw = wscale[wsi];
  float ia[8], g[8];
  #pragma unroll
  for (int v = 0; v < 8; ++v) {
    int row = m0 + v + 8 * h;
    ia[v] = invA[row] * sw;
    g[v]  = gates[(size_t)row * ED + e] * 0.5f;
  }
  #pragma unroll
  for (int t = 0; t < 4; ++t) {
    int ncol = n0 + t * 16 + l15;
    if (ncol < Nreal) {
      float b = bias[ncol];
      #pragma unroll
      for (int v = 0; v < 8; ++v) {
        int row = m0 + v + 8 * h;
        float val = (float)acc[t][v] * ia[v] + b;
        out[(size_t)row * Nreal + ncol] += g[v] * val;
      }
    }
  }
}

extern "C" void kernel_launch(void* const* d_in, const int* in_sizes, int n_in,
                              void* d_out, int out_size, void* d_ws, size_t ws_size,
                              hipStream_t stream) {
  const float* x   = (const float*)d_in[0];
  const float* Wr1 = (const float*)d_in[1];
  const float* br1 = (const float*)d_in[2];
  const float* Wr2 = (const float*)d_in[3];
  const float* br2 = (const float*)d_in[4];
  const float* W1  = (const float*)d_in[5];
  const float* b1  = (const float*)d_in[6];
  const float* W2  = (const float*)d_in[7];
  const float* b2  = (const float*)d_in[8];
  const float* W3  = (const float*)d_in[9];
  const float* b3  = (const float*)d_in[10];

  float* out   = (float*)d_out;
  float* probs = out + (size_t)BB * CD;

  char* wsp = (char*)d_ws;
  auto carve = [&](size_t bytes) -> char* {
    char* p = wsp;
    wsp += (bytes + 255) & ~(size_t)255;
    return p;
  };
  int8_t* qx     = (int8_t*)carve((size_t)BB * IND);
  float*  sx     = (float*) carve((size_t)BB * 4);
  float*  hf     = (float*) carve((size_t)BB * RHD * 4);
  float*  gates  = (float*) carve((size_t)BB * ED * 4);
  float*  wscale = (float*) carve(24 * 4);
  int8_t* qW1    = (int8_t*)carve((size_t)ED * H1D * IND);
  int8_t* qW2    = (int8_t*)carve((size_t)ED * H2D * H1D);
  int8_t* qW3    = (int8_t*)carve((size_t)ED * CPAD * H2D);
  float*  yf     = (float*) carve((size_t)BB * H1D * 4);
  int8_t* y1q    = (int8_t*)carve((size_t)BB * H1D);
  float*  s1     = (float*) carve((size_t)BB * 4);
  int8_t* y2q    = (int8_t*)carve((size_t)BB * H2D);
  float*  s2     = (float*) carve((size_t)BB * 4);

  {
    size_t n = (size_t)out_size;
    int blocks = (int)((n + 4 * 256 - 1) / (4 * 256));
    zero_kernel<<<blocks, 256, 0, stream>>>(out, n);
  }

  router_h_kernel<<<dim3(RHD / 16, BB / 16), 256, 0, stream>>>(x, Wr1, br1, hf);
  router_gate_kernel<<<BB / 8, 256, 0, stream>>>(hf, Wr2, br2, probs, gates);

  rowquant_kernel<IND><<<BB, 256, 0, stream>>>(x, qx, sx);

  wabsmean_kernel<<<24, 256, 0, stream>>>(W1, W2, W3, wscale);
  {
    size_t n1 = (size_t)ED * H1D * IND;
    size_t n2 = (size_t)ED * H2D * H1D;
    size_t n3 = (size_t)ED * CPAD * H2D;
    wquant_kernel<<<(int)(n1 / 4 / 256), 256, 0, stream>>>(W1, wscale, 0, qW1, (size_t)H1D * IND);
    wquant_kernel<<<(int)(n2 / 4 / 256), 256, 0, stream>>>(W2, wscale, 8, qW2, (size_t)H2D * H1D);
    wquant3_kernel<<<(int)(n3 / 4 / 256), 256, 0, stream>>>(W3, wscale, qW3);
  }

  for (int e = 0; e < ED; ++e) {
    gemm_iu8_relu_kernel<<<dim3(H1D / 512, BB / 16), 256, 0, stream>>>(
        qx, sx, qW1 + (size_t)e * H1D * IND, wscale, e, b1 + (size_t)e * H1D,
        yf, H1D, IND);
    rowquant_kernel<H1D><<<BB, 256, 0, stream>>>(yf, y1q, s1);

    gemm_iu8_relu_kernel<<<dim3(H2D / 512, BB / 16), 256, 0, stream>>>(
        y1q, s1, qW2 + (size_t)e * H2D * H1D, wscale, 8 + e, b2 + (size_t)e * H2D,
        yf, H2D, H1D);
    rowquant_kernel<H2D><<<BB, 256, 0, stream>>>(yf, y2q, s2);

    gemm3_iu8_gate_kernel<<<dim3(CPAD / 512, BB / 16), 256, 0, stream>>>(
        y2q, s2, qW3 + (size_t)e * CPAD * H2D, wscale, 16 + e, b3 + (size_t)e * CD,
        gates, e, out, CPAD, CD, H2D);
  }
}
